// SimpleKVcacheAttention_69174743270109
// MI455X (gfx1250) — compile-verified, matrix operand order checked by reading, not run on hardware
//
#include <hip/hip_runtime.h>
#include <hip/hip_bf16.h>

// ---------------------------------------------------------------------------
// Problem constants (from reference): E=2048, H=16, D=128, B=4, T=1024, TP=3072
// ---------------------------------------------------------------------------
#define EE   2048
#define HH   16
#define DD   128
#define BB   4
#define TT_Q 1024
#define TPP  3072
#define TTK  4096            // TP + T
#define M4   4096            // B*T

typedef __attribute__((ext_vector_type(16))) _Float16     v16h;
typedef __attribute__((ext_vector_type(8)))  _Float16     v8h;
typedef __attribute__((ext_vector_type(8)))  float        v8f;
typedef __attribute__((ext_vector_type(4)))  unsigned int u32x4;
typedef __attribute__((ext_vector_type(8)))  unsigned int u32x8;

// ---------------------------------------------------------------------------
// WMMA 16-bit A/B fragment loader (K-major rows).
// Per CDNA5 ISA 7.12.2: lanes 0-15 hold K={k0..k0+7, k0+16..k0+23},
// lanes 16-31 hold K={k0+8..k0+15, k0+24..k0+31}; row/col index = lane & 15.
// Caller passes rowptr = base + (row-for-this-lane)*ld + k0.
// ---------------------------------------------------------------------------
__device__ __forceinline__ v16h load_frag(const _Float16* rowptr, int lane) {
    const int kb = (lane & 16) >> 1;            // 0 or 8
    v8h lo = *(const v8h*)(rowptr + kb);
    v8h hi = *(const v8h*)(rowptr + kb + 16);
    v16h r;
#pragma unroll
    for (int i = 0; i < 8; ++i) { r[i] = lo[i]; r[i + 8] = hi[i]; }
    return r;
}

#define WMMA_F16(a, b, c) \
    __builtin_amdgcn_wmma_f32_16x16x32_f16(false, (a), false, (b), (short)0, (c), false, false)

// ---------------------------------------------------------------------------
// Tensor Data Mover: DMA a 2-D f16 tile (tile0 x tile1 elems, row stride
// stride0 elems) from global memory into LDS at lds_off, packed contiguously.
// D# layout per CDNA5 ISA §8.3/§8.4 (group0 = 4 SGPRs, group1 = 8 SGPRs).
// Issued per-wave; caller must gate to a single wave via a *scalar* branch
// (TDM ignores EXEC) and synchronize with s_wait_tensorcnt.
// ---------------------------------------------------------------------------
__device__ __forceinline__ void tdm_load_2d(unsigned int lds_off,
                                            const void* gptr,
                                            unsigned int tile0,
                                            unsigned int tile1,
                                            unsigned long long stride0) {
    const unsigned long long ga = (unsigned long long)(uintptr_t)gptr;
    // group0: count=1 | lds_addr | global_addr[56:0] | type=2
    u32x4 g0;
    g0[0] = 1u;
    g0[1] = lds_off;
    g0[2] = (unsigned int)ga;
    g0[3] = (unsigned int)((ga >> 32) & 0x1FFFFFFu) | (2u << 30);
    // group1: mask=0, data_size=1(2B); tensor_dim = tile dims (no OOB);
    //         tile_dim0/1 2-D tile; tensor_dim0_stride = stride0
    u32x8 g1;
    g1[0] = (1u << 16);
    g1[1] = (tile0 & 0xFFFFu) << 16;                              // tensor_dim0 lo
    g1[2] = ((tile0 >> 16) & 0xFFFFu) | ((tile1 & 0xFFFFu) << 16); // td0 hi | td1 lo
    g1[3] = ((tile1 >> 16) & 0xFFFFu) | ((tile0 & 0xFFFFu) << 16); // td1 hi | tile_dim0
    g1[4] = (tile1 & 0xFFFFu);                                     // tile_dim1 | tile_dim2=0
    g1[5] = (unsigned int)stride0;                                 // stride0[31:0]
    g1[6] = (unsigned int)((stride0 >> 32) & 0xFFFFu);             // stride0[47:32]
    g1[7] = 0u;
    asm volatile("tensor_load_to_lds %0, %1" :: "s"(g0), "s"(g1) : "memory");
}

// ---------------------------------------------------------------------------
// Kernel 1: f32 -> f16 convert (grid-stride)
// ---------------------------------------------------------------------------
__global__ void cvt_f32_f16_kernel(const float* __restrict__ s,
                                   _Float16* __restrict__ d, int n) {
    int i = blockIdx.x * blockDim.x + threadIdx.x;
    int stride = gridDim.x * blockDim.x;
    for (; i < n; i += stride) d[i] = (_Float16)s[i];
}

// ---------------------------------------------------------------------------
// Kernel 2: copy past KV cache into output cache (f32) + f16 working copies.
// transposed=0 : dsth[bh][t][d]   (K)
// transposed=1 : dsth[bh][d][t]   (V^T, so PV WMMA B-fragments are K-major)
// ---------------------------------------------------------------------------
__global__ void cache_copy_kernel(const float* __restrict__ past,
                                  float* __restrict__ outf,
                                  _Float16* __restrict__ dsth,
                                  int transposed, int n) {
    int i = blockIdx.x * blockDim.x + threadIdx.x;
    int stride = gridDim.x * blockDim.x;
    for (; i < n; i += stride) {
        int d  = i & (DD - 1);
        int t  = (i >> 7) % TPP;
        int bh = i / (TPP * DD);
        float v = past[i];
        size_t oidx = ((size_t)bh * TTK + t) * DD + d;
        outf[oidx] = v;
        if (!transposed)
            dsth[oidx] = (_Float16)v;
        else
            dsth[((size_t)bh * DD + d) * TTK + t] = (_Float16)v;
    }
}

// ---------------------------------------------------------------------------
// Kernel 3: WMMA GEMM  C = A(f16,[M,K]) @ W^T(f16,[N,K]) + bias, M=4096,N=K=2048
// Block: 256 thr = 8 waves as 4(M) x 2(N); wave tile 32x32 (2x2 WMMA tiles).
// mode 0: q -> Qh f16 [bh][t][d]
// mode 1: k -> outK f32 + Kh f16 at [bh][TP+t][d]
// mode 2: v -> outV f32 + Vth f16 at [bh][d][TP+t]
// mode 3: o -> Out f32 [row][col]
// ---------------------------------------------------------------------------
__global__ void __launch_bounds__(256)
gemm_kernel(const _Float16* __restrict__ A,
            const _Float16* __restrict__ Wq, const _Float16* __restrict__ Wk,
            const _Float16* __restrict__ Wv, const _Float16* __restrict__ Wo,
            const float* __restrict__ bq, const float* __restrict__ bk,
            const float* __restrict__ bv, const float* __restrict__ bo,
            _Float16* __restrict__ Qh,
            float* __restrict__ outK, _Float16* __restrict__ Kh,
            float* __restrict__ outV, _Float16* __restrict__ Vth,
            float* __restrict__ Out, int mode_base) {
    const int mode = mode_base + blockIdx.z;
    const _Float16* Bw;
    const float* bias;
    switch (mode) {
        case 0:  Bw = Wq; bias = bq; break;
        case 1:  Bw = Wk; bias = bk; break;
        case 2:  Bw = Wv; bias = bv; break;
        default: Bw = Wo; bias = bo; break;
    }
    const int lane = threadIdx.x & 31;
    const int w    = threadIdx.x >> 5;
    const int wm   = w >> 1;
    const int wn   = w & 1;
    const int m0   = blockIdx.y * 128 + wm * 32;
    const int n0   = blockIdx.x * 64 + wn * 32;
    const int lrow = lane & 15;

    v8f cacc[2][2];
    const v8f vzero = {};
#pragma unroll
    for (int i = 0; i < 2; ++i)
#pragma unroll
        for (int j = 0; j < 2; ++j) cacc[i][j] = vzero;

    const _Float16* arow = A  + (size_t)(m0 + lrow) * EE;
    const _Float16* brow = Bw + (size_t)(n0 + lrow) * EE;

    for (int kk = 0; kk < EE; kk += 32) {
        __builtin_prefetch(arow + kk + 512, 0, 1);   // global_prefetch_b8
        __builtin_prefetch(brow + kk + 512, 0, 1);
        v16h a0 = load_frag(arow + kk, lane);
        v16h a1 = load_frag(arow + (size_t)16 * EE + kk, lane);
        v16h b0 = load_frag(brow + kk, lane);
        v16h b1 = load_frag(brow + (size_t)16 * EE + kk, lane);
        cacc[0][0] = WMMA_F16(a0, b0, cacc[0][0]);
        cacc[0][1] = WMMA_F16(a0, b1, cacc[0][1]);
        cacc[1][0] = WMMA_F16(a1, b0, cacc[1][0]);
        cacc[1][1] = WMMA_F16(a1, b1, cacc[1][1]);
    }

    // Epilogue: C layout = VGPR r, lane -> row r + 8*(lane>=16), col lane&15
    const int cm = (lane & 16) >> 1;
    const int cn = lane & 15;
#pragma unroll
    for (int ti = 0; ti < 2; ++ti) {
#pragma unroll
        for (int tj = 0; tj < 2; ++tj) {
            const int col = n0 + tj * 16 + cn;
            const float bval = bias[col];
#pragma unroll
            for (int r = 0; r < 8; ++r) {
                const int row = m0 + ti * 16 + r + cm;
                const float val = cacc[ti][tj][r] + bval;
                if (mode == 3) {
                    Out[(size_t)row * EE + col] = val;
                } else {
                    const int b = row >> 10, t = row & (TT_Q - 1);
                    const int h = col >> 7,  d = col & (DD - 1);
                    const int bh = b * HH + h;
                    if (mode == 0) {
                        Qh[((size_t)bh * TT_Q + t) * DD + d] = (_Float16)val;
                    } else if (mode == 1) {
                        const size_t o = ((size_t)bh * TTK + TPP + t) * DD + d;
                        outK[o] = val;
                        Kh[o]   = (_Float16)val;
                    } else {
                        outV[((size_t)bh * TTK + TPP + t) * DD + d] = val;
                        Vth[((size_t)bh * DD + d) * TTK + TPP + t]  = (_Float16)val;
                    }
                }
            }
        }
    }
}

// ---------------------------------------------------------------------------
// Kernel 4: flash attention. grid = (T/128, B*H); block = 256 (8 waves).
// Each wave owns 16 query rows over D=128 (8 f32 accum tiles).
// 32-key chunks double-buffered in LDS via the Tensor Data Mover (wave 0
// issues tensor_load_to_lds, synchronized with s_wait_tensorcnt), online
// softmax with cross-lane reductions, P round-tripped through LDS.
// ---------------------------------------------------------------------------
__global__ void __launch_bounds__(256)
attn_kernel(const _Float16* __restrict__ Qh, const _Float16* __restrict__ Kh,
            const _Float16* __restrict__ Vth, _Float16* __restrict__ attnH) {
    __shared__ __align__(16) _Float16 sK[2][32 * 128];     // keys x D (x2 buf)
    __shared__ __align__(16) _Float16 sV[2][128 * 32];     // D x keys (V^T)
    __shared__ __align__(16) _Float16 sP[8 * 16 * 32];     // per-wave P tile

    const int tid  = threadIdx.x;
    const int lane = tid & 31;
    const int w    = tid >> 5;
    const int wuni = __builtin_amdgcn_readfirstlane(tid >> 5); // scalar wave id
    const int bh   = blockIdx.y;
    const int q0   = blockIdx.x * 128 + w * 16;
    const int lrow = lane & 15;
    const int cm   = (lane & 16) >> 1;
    const int cn   = lane & 15;
    const size_t kvbase = (size_t)bh * TTK * DD;           // also = bh*DD*TTK

    const _Float16* Kbase = Kh  + kvbase;                  // [4096][128]
    const _Float16* Vbase = Vth + kvbase;                  // [128][4096]

    // Load Q fragments once, folding in 1/sqrt(D)
    const _Float16* qrowp = Qh + ((size_t)bh * TT_Q + q0 + lrow) * DD;
    v16h qf[4];
#pragma unroll
    for (int ks = 0; ks < 4; ++ks) {
        qf[ks] = load_frag(qrowp + ks * 32, lane);
#pragma unroll
        for (int i = 0; i < 16; ++i)
            qf[ks][i] = qf[ks][i] * (_Float16)0.08838834764831845f;
    }

    const v8f vzero = {};
    v8f o[8];
#pragma unroll
    for (int dt = 0; dt < 8; ++dt) o[dt] = vzero;
    float mrun[8], lrun[8];
#pragma unroll
    for (int r = 0; r < 8; ++r) { mrun[r] = -1e30f; lrun[r] = 0.0f; }

    _Float16* pw = sP + w * (16 * 32);

    // Prime the pipeline: TDM pair for chunk 0 into buffer 0.
    if (wuni == 0) {
        tdm_load_2d((unsigned int)(uintptr_t)&sK[0][0], Kbase, 128u, 32u, 128ull);
        tdm_load_2d((unsigned int)(uintptr_t)&sV[0][0], Vbase, 32u, 128u,
                    (unsigned long long)TTK);
    }

    for (int kc = 0; kc < TTK; kc += 32) {
        const int buf = (kc >> 5) & 1;
        __syncthreads();   // all waves done with buf^1 -> safe to refill it
        if (wuni == 0) {
            if (kc + 32 < TTK) {
                tdm_load_2d((unsigned int)(uintptr_t)&sK[buf ^ 1][0],
                            Kbase + (size_t)(kc + 32) * DD, 128u, 32u, 128ull);
                tdm_load_2d((unsigned int)(uintptr_t)&sV[buf ^ 1][0],
                            Vbase + (kc + 32), 32u, 128u,
                            (unsigned long long)TTK);
                __builtin_amdgcn_s_wait_tensorcnt(2);  // current pair complete
            } else {
                __builtin_amdgcn_s_wait_tensorcnt(0);
            }
        }
        __syncthreads();   // current buffer visible to all waves

        const _Float16* kbuf = &sK[buf][0];
        const _Float16* vbuf = &sV[buf][0];

        // Scores: S[16 x 32] = Q[16 x 128] @ K_chunk^T
        v8f s0 = vzero, s1 = vzero;
#pragma unroll
        for (int ks = 0; ks < 4; ++ks) {
            v16h b0 = load_frag(kbuf + (size_t)lrow * 128 + ks * 32, lane);
            v16h b1 = load_frag(kbuf + (size_t)(16 + lrow) * 128 + ks * 32, lane);
            s0 = WMMA_F16(qf[ks], b0, s0);
            s1 = WMMA_F16(qf[ks], b1, s1);
        }

        // Online softmax (rows live across 16-lane halves; xor masks 1..8 stay
        // inside a half, matching C layout M = r + 8*(lane>=16), N = lane&15)
#pragma unroll
        for (int r = 0; r < 8; ++r) {
            float mx = fmaxf(s0[r], s1[r]);
            mx = fmaxf(mx, __shfl_xor(mx, 1, 32));
            mx = fmaxf(mx, __shfl_xor(mx, 2, 32));
            mx = fmaxf(mx, __shfl_xor(mx, 4, 32));
            mx = fmaxf(mx, __shfl_xor(mx, 8, 32));
            const float mn = fmaxf(mrun[r], mx);
            const float al = __expf(mrun[r] - mn);
            mrun[r] = mn;
            const float p0 = __expf(s0[r] - mn);
            const float p1 = __expf(s1[r] - mn);
            float rs = p0 + p1;
            rs += __shfl_xor(rs, 1, 32);
            rs += __shfl_xor(rs, 2, 32);
            rs += __shfl_xor(rs, 4, 32);
            rs += __shfl_xor(rs, 8, 32);
            lrun[r] = lrun[r] * al + rs;
            pw[(r + cm) * 32 + cn]      = (_Float16)p0;
            pw[(r + cm) * 32 + 16 + cn] = (_Float16)p1;
#pragma unroll
            for (int dt = 0; dt < 8; ++dt) o[dt][r] *= al;
        }

        // O += P[16x32] @ V_chunk[32x128]   (B-fragments from V^T, K-major)
        v16h pa = load_frag(pw + lrow * 32, lane);
#pragma unroll
        for (int dt = 0; dt < 8; ++dt) {
            v16h bv = load_frag(vbuf + (dt * 16 + lrow) * 32, lane);
            o[dt] = WMMA_F16(pa, bv, o[dt]);
        }
    }

    // Normalize and write attn (pre-Wo) as f16: attnH[b*T+t][h*128 + d]
    const int b = bh >> 4, h = bh & 15;
#pragma unroll
    for (int dt = 0; dt < 8; ++dt) {
#pragma unroll
        for (int r = 0; r < 8; ++r) {
            const int tq = q0 + r + cm;
            const float val = o[dt][r] / lrun[r];
            attnH[((size_t)(b * TT_Q + tq)) * EE + h * DD + dt * 16 + cn] =
                (_Float16)val;
        }
    }
}

// ---------------------------------------------------------------------------
// Host-side orchestration
// ---------------------------------------------------------------------------
extern "C" void kernel_launch(void* const* d_in, const int* in_sizes, int n_in,
                              void* d_out, int out_size, void* d_ws, size_t ws_size,
                              hipStream_t stream) {
    (void)in_sizes; (void)n_in; (void)out_size; (void)ws_size;

    const float* x      = (const float*)d_in[0];
    const float* past_k = (const float*)d_in[1];
    const float* past_v = (const float*)d_in[2];
    const float* Wq = (const float*)d_in[3]; const float* bq = (const float*)d_in[4];
    const float* Wk = (const float*)d_in[5]; const float* bk = (const float*)d_in[6];
    const float* Wv = (const float*)d_in[7]; const float* bv = (const float*)d_in[8];
    const float* Wo = (const float*)d_in[9]; const float* bo = (const float*)d_in[10];

    float* out  = (float*)d_out;                          // [4096, 2048]
    float* outK = out  + (size_t)M4 * EE;                 // [64, 4096, 128]
    float* outV = outK + (size_t)BB * HH * TTK * DD;      // [64, 4096, 128]

    // Workspace carve (f16): ~208 MB total
    _Float16* xh    = (_Float16*)d_ws;
    _Float16* Wqh   = xh   + (size_t)M4 * EE;             // 8.39M
    _Float16* Wkh   = Wqh  + (size_t)EE * EE;             // 4.19M each
    _Float16* Wvh   = Wkh  + (size_t)EE * EE;
    _Float16* Woh   = Wvh  + (size_t)EE * EE;
    _Float16* Qh    = Woh  + (size_t)EE * EE;             // 8.39M
    _Float16* Kh    = Qh   + (size_t)BB * HH * TT_Q * DD; // 33.55M
    _Float16* Vth   = Kh   + (size_t)BB * HH * TTK * DD;  // 33.55M
    _Float16* attnH = Vth  + (size_t)BB * HH * TTK * DD;  // 8.39M

    // 1) f32 -> f16 conversions
    cvt_f32_f16_kernel<<<4096, 256, 0, stream>>>(x,  xh,  M4 * EE);
    cvt_f32_f16_kernel<<<4096, 256, 0, stream>>>(Wq, Wqh, EE * EE);
    cvt_f32_f16_kernel<<<4096, 256, 0, stream>>>(Wk, Wkh, EE * EE);
    cvt_f32_f16_kernel<<<4096, 256, 0, stream>>>(Wv, Wvh, EE * EE);
    cvt_f32_f16_kernel<<<4096, 256, 0, stream>>>(Wo, Woh, EE * EE);

    // 2) past KV -> output cache sections + f16 working copies
    const int ncache = BB * HH * TPP * DD;                // 25,165,824
    cache_copy_kernel<<<8192, 256, 0, stream>>>(past_k, outK, Kh,  0, ncache);
    cache_copy_kernel<<<8192, 256, 0, stream>>>(past_v, outV, Vth, 1, ncache);

    // 3) QKV projections (mode = blockIdx.z in {0,1,2})
    gemm_kernel<<<dim3(EE / 64, M4 / 128, 3), 256, 0, stream>>>(
        xh, Wqh, Wkh, Wvh, Woh, bq, bk, bv, bo,
        Qh, outK, Kh, outV, Vth, out, 0);

    // 4) Flash attention -> attnH (f16), KV staged by the Tensor Data Mover
    attn_kernel<<<dim3(TT_Q / 128, BB * HH), 256, 0, stream>>>(Qh, Kh, Vth, attnH);

    // 5) Output projection (mode 3)
    gemm_kernel<<<dim3(EE / 64, M4 / 128, 1), 256, 0, stream>>>(
        attnH, Wqh, Wkh, Wvh, Woh, bq, bk, bv, bo,
        Qh, outK, Kh, outV, Vth, out, 3);
}
